// SelfAttentionBlock_41326175322730
// MI455X (gfx1250) — matrix operand order repaired and checked by reading, not compiled, not run on hardware
//
#include <hip/hip_runtime.h>
#include <hip/hip_bf16.h>

typedef __attribute__((ext_vector_type(16))) __bf16 v16bf;
typedef __attribute__((ext_vector_type(8)))  float  v8f;

#define EMBED  512
#define HIDDEN 2048
#define HEADS  8
#define DH     64
#define NBATCH 2
#define SEQ    2048
#define ROWS   (NBATCH * SEQ)   // 4096 token rows

#define NEG_BIG (-1.0e30f)

__device__ __forceinline__ float gelu_f(float x) {
    float x3 = x * x * x;
    return 0.5f * x * (1.0f + tanhf(0.7978845608028654f * (x + 0.044715f * x3)));
}

// A fragment (16x32 bf16, row-major with stride ld), contiguous K order:
// lanes 0-15 : row m=lane,    K = 0..15
// lanes 16-31: row m=lane-16, K = 16..31
__device__ __forceinline__ v16bf load_a_frag(const __bf16* base, int ld) {
    int lane = threadIdx.x & 31;
    int m  = lane & 15;
    int kb = (lane >> 4) << 4;
    const __bf16* p = base + m * ld + kb;
    v16bf a;
#pragma unroll
    for (int j = 0; j < 16; ++j) a[j] = p[j];
    return a;
}

// B fragment (32x16 bf16) from N-major (transposed) storage: elem(k,n)=base[n*ld+k]
// lanes 0-15 hold K=0..15 for col n=lane; lanes 16-31 hold K=16..31
__device__ __forceinline__ v16bf load_b_frag_t(const __bf16* base, int ld) {
    int lane = threadIdx.x & 31;
    int n  = lane & 15;
    int kb = (lane >> 4) << 4;
    const __bf16* p = base + n * ld + kb;
    v16bf b;
#pragma unroll
    for (int j = 0; j < 16; ++j) b[j] = p[j];
    return b;
}

__global__ void cvt_f32_bf16(const float* __restrict__ in, __bf16* __restrict__ out, int n) {
    int i = blockIdx.x * 256 + threadIdx.x;
    if (i < n) out[i] = (__bf16)in[i];
}

// f32 [R][C] -> bf16 [C][R] (weights: convert + transpose in one pass)
__global__ __launch_bounds__(256)
void cvt_transpose_f32_bf16(const float* __restrict__ in, __bf16* __restrict__ out,
                            int R, int C) {
    __shared__ __bf16 tile[32][33];
    const int c0 = blockIdx.x * 32, r0 = blockIdx.y * 32;
    const int tx = threadIdx.x, ty = threadIdx.y;   // 32 x 8
#pragma unroll
    for (int i = 0; i < 32; i += 8)
        tile[ty + i][tx] = (__bf16)in[(size_t)(r0 + ty + i) * C + c0 + tx];
    __syncthreads();
#pragma unroll
    for (int i = 0; i < 32; i += 8)
        out[(size_t)(c0 + ty + i) * R + r0 + tx] = tile[tx][ty + i];
}

// bf16 [NBATCH][SEQ][EMBED] -> bf16 [NBATCH][EMBED][SEQ]  (V pre-transpose)
__global__ __launch_bounds__(256)
void transpose_b16_batched(const __bf16* __restrict__ in, __bf16* __restrict__ out) {
    __shared__ __bf16 tile[32][33];
    const int nb = blockIdx.z;
    const int s0 = blockIdx.x * 32, c0 = blockIdx.y * 32;
    const int tx = threadIdx.x, ty = threadIdx.y;   // 32 x 8
#pragma unroll
    for (int i = 0; i < 32; i += 8)
        tile[ty + i][tx] = in[(size_t)(nb * SEQ + s0 + ty + i) * EMBED + c0 + tx];
    __syncthreads();
#pragma unroll
    for (int i = 0; i < 32; i += 8)
        out[(size_t)nb * EMBED * SEQ + (size_t)(c0 + ty + i) * SEQ + s0 + tx] = tile[tx][ty + i];
}

// C[M,N] = act(A[M,K] @ W[K,N] + bias[N]) -> bf16, with W supplied TRANSPOSED (Wt[N][K]).
// Block: 256 threads (8 waves). Tile BM=128, BN=128, BK=32, double-buffered LDS.
// Wave grid 4x2, each wave computes a 32x64 tile = 8 WMMA accumulators.
template <int ACT>
__global__ __launch_bounds__(256)
void gemm_bias_act(const __bf16* __restrict__ A, const __bf16* __restrict__ Wt,
                   const float* __restrict__ bias, __bf16* __restrict__ out,
                   int M, int N, int K) {
    __shared__ __bf16 sA[2][128 * 32];    // row-major, ld=32 (8KB per stage)
    __shared__ __bf16 sBt[2][128 * 32];   // N-major [n][k], ld=32 (8KB per stage)

    const int t       = threadIdx.x;
    const int wave    = t >> 5;
    const int waveRow = wave >> 1;      // 0..3 -> 32-row band
    const int waveCol = wave & 1;       // 0..1 -> 64-col band
    const int row0    = blockIdx.y * 128;
    const int col0    = blockIdx.x * 128;

    // Global-load mappings (per thread)
    const int ar  = t >> 2;             // 0..63  (A rows ar and ar+64)
    const int ac  = (t & 3) << 3;       // 0,8,16,24
    const int bn  = t >> 1;             // 0..127 (Wt row = output column)
    const int bks = (t & 1) << 4;       // 0 or 16

    // Prologue: stage K-tile 0 through registers into LDS stage 0
    uint4 aReg0 = *(const uint4*)(A + (size_t)(row0 + ar) * K + ac);
    uint4 aReg1 = *(const uint4*)(A + (size_t)(row0 + ar + 64) * K + ac);
    uint4 bReg0 = *(const uint4*)(Wt + (size_t)(col0 + bn) * K + bks);
    uint4 bReg1 = *(const uint4*)(Wt + (size_t)(col0 + bn) * K + bks + 8);
    *(uint4*)(&sA[0][ar * 32 + ac])         = aReg0;
    *(uint4*)(&sA[0][(ar + 64) * 32 + ac])  = aReg1;
    *(uint4*)(&sBt[0][bn * 32 + bks])       = bReg0;
    *(uint4*)(&sBt[0][bn * 32 + bks + 8])   = bReg1;

    const v8f z8 = {0.f, 0.f, 0.f, 0.f, 0.f, 0.f, 0.f, 0.f};
    v8f acc[2][4] = {{z8, z8, z8, z8}, {z8, z8, z8, z8}};

    const int nk = K >> 5;
    for (int it = 0; it < nk; ++it) {
        __syncthreads();
        const int  cur  = it & 1;
        const bool more = (it + 1 < nk);
        if (more) {  // issue next tile's global loads before compute (latency hiding)
            const int k0 = (it + 1) << 5;
            aReg0 = *(const uint4*)(A + (size_t)(row0 + ar) * K + k0 + ac);
            aReg1 = *(const uint4*)(A + (size_t)(row0 + ar + 64) * K + k0 + ac);
            bReg0 = *(const uint4*)(Wt + (size_t)(col0 + bn) * K + k0 + bks);
            bReg1 = *(const uint4*)(Wt + (size_t)(col0 + bn) * K + k0 + bks + 8);
        }

        v16bf af[2], bfr[4];
#pragma unroll
        for (int i = 0; i < 2; ++i)
            af[i] = load_a_frag(&sA[cur][(waveRow * 32 + i * 16) * 32], 32);
#pragma unroll
        for (int j = 0; j < 4; ++j)
            bfr[j] = load_b_frag_t(&sBt[cur][(waveCol * 64 + j * 16) * 32], 32);
#pragma unroll
        for (int i = 0; i < 2; ++i)
#pragma unroll
            for (int j = 0; j < 4; ++j)
                acc[i][j] = __builtin_amdgcn_wmma_f32_16x16x32_bf16(
                    false, af[i], false, bfr[j], (short)0, acc[i][j], false, false);

        if (more) {  // drain staged registers into the other LDS buffer
            const int nxt = cur ^ 1;
            *(uint4*)(&sA[nxt][ar * 32 + ac])        = aReg0;
            *(uint4*)(&sA[nxt][(ar + 64) * 32 + ac]) = aReg1;
            *(uint4*)(&sBt[nxt][bn * 32 + bks])      = bReg0;
            *(uint4*)(&sBt[nxt][bn * 32 + bks + 8])  = bReg1;
        }
    }

    // Epilogue: bias + activation, store bf16.
    const int lane = t & 31;
    const int half = lane >> 4;
    const int nl   = lane & 15;
#pragma unroll
    for (int i = 0; i < 2; ++i)
#pragma unroll
        for (int j = 0; j < 4; ++j) {
            int colg = col0 + waveCol * 64 + j * 16 + nl;
            float bv = bias[colg];
#pragma unroll
            for (int r = 0; r < 8; ++r) {
                int rowg = row0 + waveRow * 32 + i * 16 + r + 8 * half;
                float vv = acc[i][j][r] + bv;
                if (ACT) vv = gelu_f(vv);
                out[(size_t)rowg * N + colg] = (__bf16)vv;
            }
        }
}

// Flash attention, causal. One wave per (batch, head, 16-query tile); 64-key chunks.
// Q,K: bf16 [ROWS][EMBED] (head h at cols h*DH..). Vt: bf16 [NBATCH][EMBED][SEQ]
// (pre-transposed so PV B-fragments are contiguous). out: f32 [ROWS][EMBED].
__global__ __launch_bounds__(256)
void attn_kernel(const __bf16* __restrict__ Q, const __bf16* __restrict__ Km,
                 const __bf16* __restrict__ Vt, float* __restrict__ out) {
    __shared__ __bf16 sP[8][16 * 64];  // per-wave P bounce buffer (2KB/wave)

    const int t    = threadIdx.x;
    const int wave = t >> 5;
    const int lane = t & 31;
    const int half = lane >> 4;
    const int nl   = lane & 15;

    int task  = blockIdx.x * 8 + wave;         // 2048 tasks total
    int qtile = task & 127;                    // SEQ/16 = 128
    int h     = (task >> 7) & 7;
    int nb    = task >> 10;
    const int    q0    = qtile * 16;
    const size_t base  = (size_t)nb * SEQ * EMBED + (size_t)h * DH;       // Q/K rows
    const size_t vbase = (size_t)nb * EMBED * SEQ + (size_t)h * DH * SEQ; // Vt rows

    // Q fragments: dh=64 split into two K=32 A-fragments
    v16bf qa[2];
#pragma unroll
    for (int c = 0; c < 2; ++c) {
        const __bf16* p = Q + base + (size_t)(q0 + nl) * EMBED + half * 16 + c * 32;
        v16bf a;
#pragma unroll
        for (int j = 0; j < 16; ++j) a[j] = p[j];
        qa[c] = a;
    }

    const v8f z8 = {0.f, 0.f, 0.f, 0.f, 0.f, 0.f, 0.f, 0.f};
    v8f o[4] = {z8, z8, z8, z8};
    float mrow[8], lrow[8];
#pragma unroll
    for (int r = 0; r < 8; ++r) { mrow[r] = NEG_BIG; lrow[r] = 0.f; }

    const int kend = (q0 + 15) >> 6;           // 64-key chunks up to the diagonal
    for (int kc = 0; kc <= kend; ++kc) {
        const int key0 = kc * 64;
        v8f s[4] = {z8, z8, z8, z8};
#pragma unroll
        for (int c = 0; c < 2; ++c) {
#pragma unroll
            for (int tn = 0; tn < 4; ++tn) {
                v16bf kf;
                const __bf16* p = Km + base + (size_t)(key0 + tn * 16 + nl) * EMBED + half * 16 + c * 32;
#pragma unroll
                for (int j = 0; j < 16; ++j) kf[j] = p[j];
                s[tn] = __builtin_amdgcn_wmma_f32_16x16x32_bf16(
                    false, qa[c], false, kf, (short)0, s[tn], false, false);
            }
        }
        const bool need_mask = (kc == kend);
#pragma unroll
        for (int r = 0; r < 8; ++r) {
            int   qidx = q0 + r + 8 * half;
            float v[4];
#pragma unroll
            for (int tn = 0; tn < 4; ++tn) {
                v[tn] = s[tn][r] * 0.125f;       // 1/sqrt(64)
                if (need_mask && (key0 + tn * 16 + nl > qidx)) v[tn] = NEG_BIG;
            }
            // row max across the 16 lanes of this half (xor masks < 16 stay in-half)
            float cm = fmaxf(fmaxf(v[0], v[1]), fmaxf(v[2], v[3]));
#pragma unroll
            for (int msk = 1; msk < 16; msk <<= 1) cm = fmaxf(cm, __shfl_xor(cm, msk, 32));
            float nm    = fmaxf(mrow[r], cm);
            float alpha = __expf(mrow[r] - nm);
            float pr[4];
            float rs = 0.f;
#pragma unroll
            for (int tn = 0; tn < 4; ++tn) { pr[tn] = __expf(v[tn] - nm); rs += pr[tn]; }
#pragma unroll
            for (int msk = 1; msk < 16; msk <<= 1) rs += __shfl_xor(rs, msk, 32);
            lrow[r] = lrow[r] * alpha + rs;
            mrow[r] = nm;
#pragma unroll
            for (int u = 0; u < 4; ++u) o[u][r] *= alpha;
            int m = r + 8 * half;
#pragma unroll
            for (int tn = 0; tn < 4; ++tn)
                sP[wave][m * 64 + tn * 16 + nl] = (__bf16)pr[tn];
        }
        __builtin_amdgcn_wave_barrier();
        // Re-read P from LDS in A-fragment layout: two K=32 fragments over 64 keys
        v16bf pa[2];
#pragma unroll
        for (int c = 0; c < 2; ++c) {
            const __bf16* p = &sP[wave][nl * 64 + c * 32 + half * 16];
#pragma unroll
            for (int j = 0; j < 16; ++j) pa[c][j] = p[j];
        }
        __builtin_amdgcn_wave_barrier();
        // P @ V: four 16-col output tiles over dh, two K=32 chunks over keys.
        // Vt is d-major, so each B-fragment is 16 contiguous bf16 per lane.
#pragma unroll
        for (int u = 0; u < 4; ++u) {
#pragma unroll
            for (int c = 0; c < 2; ++c) {
                v16bf vf;
                const __bf16* p = Vt + vbase + (size_t)(u * 16 + nl) * SEQ + key0 + c * 32 + half * 16;
#pragma unroll
                for (int j = 0; j < 16; ++j) vf[j] = p[j];
                o[u] = __builtin_amdgcn_wmma_f32_16x16x32_bf16(
                    false, pa[c], false, vf, (short)0, o[u], false, false);
            }
        }
    }

    // Normalize and store f32 output in [n, s, h*DH + d] layout
#pragma unroll
    for (int u = 0; u < 4; ++u)
#pragma unroll
        for (int r = 0; r < 8; ++r) {
            int m = r + 8 * half;
            out[(size_t)nb * SEQ * EMBED + (size_t)(q0 + m) * EMBED + h * DH + u * 16 + nl] =
                o[u][r] / lrow[r];
        }
}

extern "C" void kernel_launch(void* const* d_in, const int* in_sizes, int n_in,
                              void* d_out, int out_size, void* d_ws, size_t ws_size,
                              hipStream_t stream) {
    (void)in_sizes; (void)n_in; (void)out_size; (void)ws_size;

    const float* x   = (const float*)d_in[0];
    const float* Wq1 = (const float*)d_in[1];
    const float* bq1 = (const float*)d_in[2];
    const float* Wq2 = (const float*)d_in[3];
    const float* bq2 = (const float*)d_in[4];
    const float* Wk1 = (const float*)d_in[5];
    const float* bk1 = (const float*)d_in[6];
    const float* Wk2 = (const float*)d_in[7];
    const float* bk2 = (const float*)d_in[8];
    const float* Wv1 = (const float*)d_in[9];
    const float* bv1 = (const float*)d_in[10];
    const float* Wv2 = (const float*)d_in[11];
    const float* bv2 = (const float*)d_in[12];

    char*  ws  = (char*)d_ws;
    size_t off = 0;
    __bf16* xb   = (__bf16*)(ws + off); off += (size_t)ROWS * EMBED * 2;
    __bf16* wq1t = (__bf16*)(ws + off); off += (size_t)EMBED * HIDDEN * 2;  // [HIDDEN][EMBED]
    __bf16* wq2t = (__bf16*)(ws + off); off += (size_t)HIDDEN * EMBED * 2;  // [EMBED][HIDDEN]
    __bf16* wk1t = (__bf16*)(ws + off); off += (size_t)EMBED * HIDDEN * 2;
    __bf16* wk2t = (__bf16*)(ws + off); off += (size_t)HIDDEN * EMBED * 2;
    __bf16* wv1t = (__bf16*)(ws + off); off += (size_t)EMBED * HIDDEN * 2;
    __bf16* wv2t = (__bf16*)(ws + off); off += (size_t)HIDDEN * EMBED * 2;
    __bf16* H    = (__bf16*)(ws + off); off += (size_t)ROWS * HIDDEN * 2;   // reused q/k/v
    __bf16* qB   = (__bf16*)(ws + off); off += (size_t)ROWS * EMBED * 2;
    __bf16* kB   = (__bf16*)(ws + off); off += (size_t)ROWS * EMBED * 2;
    __bf16* vB   = (__bf16*)(ws + off); off += (size_t)ROWS * EMBED * 2;
    __bf16* vT   = (__bf16*)(ws + off); off += (size_t)ROWS * EMBED * 2;    // [nb][EMBED][SEQ]

    dim3 tb(32, 8);
    // x: plain convert; weights: convert + transpose (Wt[N][K])
    cvt_f32_bf16<<<(ROWS * EMBED + 255) / 256, 256, 0, stream>>>(x, xb, ROWS * EMBED);
    cvt_transpose_f32_bf16<<<dim3(HIDDEN / 32, EMBED / 32), tb, 0, stream>>>(Wq1, wq1t, EMBED, HIDDEN);
    cvt_transpose_f32_bf16<<<dim3(EMBED / 32, HIDDEN / 32), tb, 0, stream>>>(Wq2, wq2t, HIDDEN, EMBED);
    cvt_transpose_f32_bf16<<<dim3(HIDDEN / 32, EMBED / 32), tb, 0, stream>>>(Wk1, wk1t, EMBED, HIDDEN);
    cvt_transpose_f32_bf16<<<dim3(EMBED / 32, HIDDEN / 32), tb, 0, stream>>>(Wk2, wk2t, HIDDEN, EMBED);
    cvt_transpose_f32_bf16<<<dim3(HIDDEN / 32, EMBED / 32), tb, 0, stream>>>(Wv1, wv1t, EMBED, HIDDEN);
    cvt_transpose_f32_bf16<<<dim3(EMBED / 32, HIDDEN / 32), tb, 0, stream>>>(Wv2, wv2t, HIDDEN, EMBED);

    dim3 gL1(HIDDEN / 128, ROWS / 128);  // 16 x 32
    dim3 gL2(EMBED / 128, ROWS / 128);   //  4 x 32

    // Q = gelu(x@Wq1+bq1)@Wq2+bq2
    gemm_bias_act<1><<<gL1, 256, 0, stream>>>(xb, wq1t, bq1, H, ROWS, HIDDEN, EMBED);
    gemm_bias_act<0><<<gL2, 256, 0, stream>>>(H, wq2t, bq2, qB, ROWS, EMBED, HIDDEN);
    // K
    gemm_bias_act<1><<<gL1, 256, 0, stream>>>(xb, wk1t, bk1, H, ROWS, HIDDEN, EMBED);
    gemm_bias_act<0><<<gL2, 256, 0, stream>>>(H, wk2t, bk2, kB, ROWS, EMBED, HIDDEN);
    // V = gelu(gelu(x@Wv1+bv1)@Wv2+bv2)
    gemm_bias_act<1><<<gL1, 256, 0, stream>>>(xb, wv1t, bv1, H, ROWS, HIDDEN, EMBED);
    gemm_bias_act<1><<<gL2, 256, 0, stream>>>(H, wv2t, bv2, vB, ROWS, EMBED, HIDDEN);

    // Pre-transpose V for contiguous PV B-fragments
    transpose_b16_batched<<<dim3(SEQ / 32, EMBED / 32, NBATCH), tb, 0, stream>>>(vB, vT);

    // causal flash attention
    attn_kernel<<<(NBATCH * HEADS * (SEQ / 16)) / 8, 256, 0, stream>>>(qB, kB, vT, (float*)d_out);
}
